// FullRankRNN_59760174957212
// MI455X (gfx1250) — compile-verified, matrix operand order checked by reading, not run on hardware
//
#include <hip/hip_runtime.h>
#include <hip/hip_bf16.h>

// Problem constants (FullRankRNN reference)
#define B_ 64
#define T_ 1024
#define H_ 512
#define I_ 64
#define O_ 8
#define NOISE_STD 0.05f
#define ALPHA_ 0.2f

typedef __attribute__((ext_vector_type(16))) __bf16 v16bf;
typedef __attribute__((ext_vector_type(8)))  __bf16 v8bf;
typedef __attribute__((ext_vector_type(8)))  float   v8f;

static constexpr int RS = 520;   // LDS row stride for the relu(h) tile (bf16 elems)

// K mapping inside a 16x32 bf16 A fragment / 32x16 B fragment (CDNA5 ISA 7.12.2):
// lanes 0-15: VGPR0..3 hold K pairs {0..7}, VGPR4..7 hold {16..23}; lanes 16-31: +8.
__device__ __forceinline__ int kmap(int e, int hiHalf) {
    int pair = e >> 1, sub = e & 1;
    int kb = (pair < 4) ? (pair * 2) : (16 + (pair - 4) * 2);
    return kb + sub + hiHalf * 8;
}

// ---------------- prep kernels (run once per launch, tiny) ----------------

// Pack |wrec|*mask into bf16 B fragments, per-lane contiguous:
// wrecB[((nt*16 + kb)*32 + lane)*16 + e]
__global__ void prep_wrecB(const float* __restrict__ wrec,
                           const float* __restrict__ mask,
                           __bf16* __restrict__ wrecB) {
    int idx = blockIdx.x * blockDim.x + threadIdx.x;   // 32*16*32*16 = 262144
    if (idx >= 32 * 16 * 32 * 16) return;
    int e  = idx & 15;
    int l  = (idx >> 4) & 31;
    int kb = (idx >> 9) & 15;
    int nt = idx >> 13;
    int n = nt * 16 + (l & 15);
    int k = kb * 32 + kmap(e, l >> 4);
    float v = fabsf(wrec[k * H_ + n]) * mask[k * H_ + n];
    wrecB[idx] = (__bf16)v;
}

// Pack (wi * si)*mask into bf16 B fragments: wiB[((nt*2 + kb)*32 + lane)*16 + e]
__global__ void prep_wiB(const float* __restrict__ wi,
                         const float* __restrict__ si,
                         const float* __restrict__ mask,
                         __bf16* __restrict__ wiB) {
    int idx = blockIdx.x * blockDim.x + threadIdx.x;   // 32*2*32*16 = 32768
    if (idx >= 32 * 2 * 32 * 16) return;
    int e  = idx & 15;
    int l  = (idx >> 4) & 31;
    int kb = (idx >> 9) & 1;
    int nt = idx >> 10;
    int n = nt * 16 + (l & 15);
    int k = kb * 32 + kmap(e, l >> 4);
    float v = wi[k * H_ + n] * si[k] * mask[k * H_ + n];
    wiB[idx] = (__bf16)v;
}

// wo_eff[n][o] = wo[n][o] * so[o] * mask[n][o]  (fp32, 16KB, stays in L2)
__global__ void prep_wo(const float* __restrict__ wo,
                        const float* __restrict__ so,
                        const float* __restrict__ mask,
                        float* __restrict__ woE) {
    int idx = blockIdx.x * blockDim.x + threadIdx.x;   // H*O = 4096
    if (idx >= H_ * O_) return;
    int o = idx & (O_ - 1);
    woE[idx] = wo[idx] * so[o] * mask[idx];
}

// Build a 16x32 bf16 A fragment from fp32 (p[0..7], p[16..23])
__device__ __forceinline__ v16bf cvtA16(const float* __restrict__ p) {
    v16bf a;
#pragma unroll
    for (int e = 0; e < 8; ++e) {
        a[e]     = (__bf16)p[e];
        a[e + 8] = (__bf16)p[e + 16];
    }
    return a;
}

// A fragment (16x32) for kb from the row-major padded LDS relu(h) tile
__device__ __forceinline__ v16bf ldsAfrag(const __bf16* __restrict__ rb,
                                          int lo, int hiH, int kb) {
    const __bf16* ap = rb + lo * RS + kb * 32 + hiH * 8;
    v8bf alo = *(const v8bf*)ap;          // K 0..7 (+8 for hi half)
    v8bf ahi = *(const v8bf*)(ap + 16);   // K 16..23
    return __builtin_shufflevector(alo, ahi,
        0,1,2,3,4,5,6,7,8,9,10,11,12,13,14,15);
}

// B fragment (32x16) for (nt, kbh) from plane-split LDS (lane stride 16B)
__device__ __forceinline__ v16bf ldsBfrag(const __bf16* __restrict__ ldsB,
                                          int nt, int kbh, int lane) {
    const __bf16* bp = ldsB + ((size_t)(nt * 8 + kbh) * 2) * 256 + lane * 8;
    v8bf blo = *(const v8bf*)bp;
    v8bf bhi = *(const v8bf*)(bp + 256);
    return __builtin_shufflevector(blo, bhi,
        0,1,2,3,4,5,6,7,8,9,10,11,12,13,14,15);
}

#define WMMA_BF16(A, Bm, C) \
    __builtin_amdgcn_wmma_f32_16x16x32_bf16(false, (A), false, (Bm), (short)0, (C), false, false)

// ---------------- main sequential scan kernel ----------------
// grid = 4 workgroups (batch tiles of 16), block = 512 threads = 16 waves.
// Wave w owns h columns [32w, 32w+32) (two 16-col WMMA N-tiles).
// Weight residency: Wrec kb 0..7 in VGPRs (128/lane), kb 8..15 in LDS (256KB),
// Wi in VGPRs. relu(h) exchanged via double-buffered bf16 LDS tile.
// A and LDS-B fragments are software-pipelined one kb ahead.
__launch_bounds__(512, 1)
__global__ void rnn_scan(const float* __restrict__ input,   // [B,T,I] fp32
                         const float* __restrict__ noise,   // [B,T,H] fp32
                         const __bf16* __restrict__ wrecB,  // packed B frags
                         const __bf16* __restrict__ wiB,    // packed B frags
                         const float* __restrict__ bvec,    // [H]
                         const float* __restrict__ h0,      // [H]
                         float* __restrict__ traj) {        // [B,T+1,H]
    __shared__ __align__(16) __bf16 rbuf[2][16 * RS];          // 33.3 KB
    __shared__ __align__(16) __bf16 ldsB[32 * 8 * 2 * 32 * 8]; // 256 KB (kb 8..15)

    const int tid  = threadIdx.x;
    const int wave = tid >> 5;                // 0..15
    const int lane = tid & 31;
    const int lo   = lane & 15;
    const int hiH  = lane >> 4;
    const int b0   = blockIdx.x * 16;         // batch tile base
    const int nt0  = wave * 2;
    const int nt1  = wave * 2 + 1;
    const int n0   = wave * 32 + lo;          // D-frag column, tile 0
    const int n1   = wave * 32 + 16 + lo;     // D-frag column, tile 1

    // ---- prologue A: stage Wrec kb 8..15 into LDS (plane-split: 16B/lane/plane)
    for (int c = tid; c < 32 * 8 * 2 * 32; c += 512) {
        int l   = c & 31;
        int pl  = (c >> 5) & 1;
        int kbh = (c >> 6) & 7;
        int nt  = c >> 9;
        v8bf v = *(const v8bf*)(wrecB + (((nt * 16) + (kbh + 8)) * 32 + l) * 16 + pl * 8);
        *(v8bf*)(ldsB + (size_t)c * 8) = v;
    }

    // ---- prologue B: rbuf[0] = relu(h0); traj[:,0,:] = h0; h registers = h0[n]
    {
        const int c = wave * 32 + lane;       // one column per lane
        const float h0c = h0[c];
        const __bf16 r0 = (__bf16)fmaxf(h0c, 0.0f);
#pragma unroll 4
        for (int m = 0; m < 16; ++m) {
            rbuf[0][m * RS + c] = r0;
            traj[((long)(b0 + m) * (T_ + 1)) * H_ + c] = h0c;
        }
    }
    const float bi0 = bvec[n0];
    const float bi1 = bvec[n1];
    v8f hA, hB;
    {
        float h0a = h0[n0], h0b = h0[n1];
#pragma unroll
        for (int g = 0; g < 8; ++g) { hA[g] = h0a; hB[g] = h0b; }
    }

    // ---- prologue C: register-resident weight fragments (live over t-loop)
    v16bf Br0[8], Br1[8], Wi0[2], Wi1[2];
#pragma unroll
    for (int kb = 0; kb < 8; ++kb) {
        Br0[kb] = *(const v16bf*)(wrecB + (((nt0 * 16) + kb) * 32 + lane) * 16);
        Br1[kb] = *(const v16bf*)(wrecB + (((nt1 * 16) + kb) * 32 + lane) * 16);
    }
#pragma unroll
    for (int kb = 0; kb < 2; ++kb) {
        Wi0[kb] = *(const v16bf*)(wiB + ((nt0 * 2 + kb) * 32 + lane) * 16);
        Wi1[kb] = *(const v16bf*)(wiB + ((nt1 * 2 + kb) * 32 + lane) * 16);
    }
    __syncthreads();

    int cur = 0;
    for (int t = 0; t < T_; ++t) {
        const __bf16* __restrict__ rb = rbuf[cur];

        // ---- x_t @ Wi first: initializes accumulators, xf regs die early.
        // (x lines were prefetched by the previous step -> near-cache hit.)
        const float* xrow = input + ((long)(b0 + lo) * T_ + t) * I_;
        v8f acc0, acc1;
        {
            v16bf xf0 = cvtA16(xrow + hiH * 8);        // K 0..31 slice
            v16bf xf1 = cvtA16(xrow + 32 + hiH * 8);   // K 32..63
            v8f z = {};
            acc0 = WMMA_BF16(xf0, Wi0[0], z);
            acc0 = WMMA_BF16(xf1, Wi0[1], acc0);
            acc1 = WMMA_BF16(xf0, Wi1[0], z);
            acc1 = WMMA_BF16(xf1, Wi1[1], acc1);
        }

        // ---- r @ Wrec, kb 0..7: A from LDS (pipelined 1 deep), B registers
        v16bf aCur = ldsAfrag(rb, lo, hiH, 0);
#pragma unroll
        for (int kb = 0; kb < 8; ++kb) {
            v16bf aNxt = ldsAfrag(rb, lo, hiH, kb + 1);  // kb=7 preloads frag 8
            acc0 = WMMA_BF16(aCur, Br0[kb], acc0);
            acc1 = WMMA_BF16(aCur, Br1[kb], acc1);
            aCur = aNxt;
        }
        // ---- r @ Wrec, kb 8..15: A + B from LDS, both pipelined 1 deep
        v16bf bb0 = ldsBfrag(ldsB, nt0, 0, lane);
        v16bf bb1 = ldsBfrag(ldsB, nt1, 0, lane);
#pragma unroll
        for (int kbh = 0; kbh < 8; ++kbh) {
            v16bf aNxt = (kbh < 7) ? ldsAfrag(rb, lo, hiH, kbh + 9) : aCur;
            v16bf nb0  = (kbh < 7) ? ldsBfrag(ldsB, nt0, kbh + 1, lane) : bb0;
            v16bf nb1  = (kbh < 7) ? ldsBfrag(ldsB, nt1, kbh + 1, lane) : bb1;
            acc0 = WMMA_BF16(aCur, bb0, acc0);
            acc1 = WMMA_BF16(aCur, bb1, acc1);
            aCur = aNxt;
            bb0 = nb0;
            bb1 = nb1;
        }

        // ---- prefetch next step's streaming inputs (no VGPR cost)
        if (t + 1 < T_) {
            const float* xn = input + ((long)(b0 + lo) * T_ + (t + 1)) * I_;
            __builtin_prefetch(xn + hiH * 8, 0, 1);
            __builtin_prefetch(xn + 32 + hiH * 8, 0, 1);
            // one 128B line per (row, wave) of next noise slice
            const float* nn = noise + ((long)(b0 + lo) * T_ + (t + 1)) * H_;
            __builtin_prefetch(nn + wave * 32, 0, 1);
        }

        // ---- noise for this step (prefetched last step -> near-cache)
        float nz0[8], nz1[8];
        {
            const float* nrow = noise + ((long)(b0 + hiH * 8) * T_ + t) * H_;
#pragma unroll
            for (int g = 0; g < 8; ++g) {
                nz0[g] = nrow[(long)g * (T_ * H_) + n0];
                nz1[g] = nrow[(long)g * (T_ * H_) + n1];
            }
        }

        // ---- state update, trajectory store, publish relu(h_new) for next step
        const int nxt = cur ^ 1;
        __bf16* __restrict__ wb = rbuf[nxt];
#pragma unroll
        for (int g = 0; g < 8; ++g) {
            const int m = g + hiH * 8;                       // D-frag row
            const float hv0 = hA[g], hv1 = hB[g];
            const float hn0 = hv0 + NOISE_STD * nz0[g] + ALPHA_ * (-hv0 + acc0[g] + bi0);
            const float hn1 = hv1 + NOISE_STD * nz1[g] + ALPHA_ * (-hv1 + acc1[g] + bi1);
            const long tb = ((long)(b0 + m) * (T_ + 1) + (t + 1)) * H_;
            traj[tb + n0] = hn0;
            traj[tb + n1] = hn1;
            hA[g] = hn0;
            hB[g] = hn1;
            wb[m * RS + n0] = (__bf16)fmaxf(hn0, 0.0f);
            wb[m * RS + n1] = (__bf16)fmaxf(hn1, 0.0f);
        }
        __syncthreads();
        cur = nxt;
    }
}

// ---------------- output pass: out[b,t,:] = relu(traj[b,t+1,:]) @ wo_eff ----
__global__ void out_kernel(const float* __restrict__ traj,
                           const float* __restrict__ woE,
                           float* __restrict__ out) {
    int idx = blockIdx.x * blockDim.x + threadIdx.x;  // B*T
    if (idx >= B_ * T_) return;
    int b = idx / T_, t = idx - b * T_;
    const float* hrow = traj + ((long)b * (T_ + 1) + (t + 1)) * H_;
    float acc[O_] = {0, 0, 0, 0, 0, 0, 0, 0};
    for (int n = 0; n < H_; n += 4) {
        float4 hv = *(const float4*)(hrow + n);
        float r0 = fmaxf(hv.x, 0.0f), r1 = fmaxf(hv.y, 0.0f);
        float r2 = fmaxf(hv.z, 0.0f), r3 = fmaxf(hv.w, 0.0f);
#pragma unroll
        for (int o = 0; o < O_; ++o) {
            acc[o] += r0 * woE[(n + 0) * O_ + o] + r1 * woE[(n + 1) * O_ + o]
                    + r2 * woE[(n + 2) * O_ + o] + r3 * woE[(n + 3) * O_ + o];
        }
    }
    float* op = out + (long)idx * O_;
#pragma unroll
    for (int o = 0; o < O_; ++o) op[o] = acc[o];
}

extern "C" void kernel_launch(void* const* d_in, const int* in_sizes, int n_in,
                              void* d_out, int out_size, void* d_ws, size_t ws_size,
                              hipStream_t stream) {
    const float* input     = (const float*)d_in[0];
    const float* noise     = (const float*)d_in[1];
    const float* wi        = (const float*)d_in[2];
    const float* si        = (const float*)d_in[3];
    const float* wrec      = (const float*)d_in[4];
    const float* bvec      = (const float*)d_in[5];
    const float* wo        = (const float*)d_in[6];
    const float* so        = (const float*)d_in[7];
    const float* wi_mask   = (const float*)d_in[8];
    const float* wrec_mask = (const float*)d_in[9];
    const float* wo_mask   = (const float*)d_in[10];
    const float* h0        = (const float*)d_in[11];

    float* out  = (float*)d_out;                       // [B,T,O]
    float* traj = out + (size_t)B_ * T_ * O_;          // [B,T+1,H]

    char* ws = (char*)d_ws;
    __bf16* wrecB = (__bf16*)(ws);                     // 512 KB
    __bf16* wiB   = (__bf16*)(ws + 524288);            // 64 KB
    float*  woE   = (float*)(ws + 524288 + 65536);     // 16 KB

    prep_wrecB<<<1024, 256, 0, stream>>>(wrec, wrec_mask, wrecB);
    prep_wiB<<<128, 256, 0, stream>>>(wi, si, wi_mask, wiB);
    prep_wo<<<16, 256, 0, stream>>>(wo, so, wo_mask, woE);
    rnn_scan<<<4, 512, 0, stream>>>(input, noise, wrecB, wiB, bvec, h0, traj);
    out_kernel<<<(B_ * T_ + 255) / 256, 256, 0, stream>>>(traj, woE, out);
}
